// SpreaderPerdictor_77472620085323
// MI455X (gfx1250) — matrix operand order repaired and checked by reading, not compiled, not run on hardware
//
#include <hip/hip_runtime.h>
#include <hip/hip_bf16.h>
#include <stdint.h>

// ---------------------------------------------------------------------------
// Problem constants (match reference)
// ---------------------------------------------------------------------------
#define NU   40000
#define EU   640000
#define BB   64
#define TT   16
#define NW   30
#define ET   60
#define NT   (BB * TT * NW)    // 30720
#define NTE  (BB * TT * ET)    // 61440
#define NSEG (BB * TT)         // 1024

typedef __attribute__((ext_vector_type(16))) _Float16 v16h;
typedef __attribute__((ext_vector_type(8)))  float    v8f;

// ---------------------------------------------------------------------------
// Small utility kernels
// ---------------------------------------------------------------------------
__global__ void fill_f32_kernel(float* __restrict__ p, float v, long long n) {
    long long i = (long long)blockIdx.x * blockDim.x + threadIdx.x;
    if (i < n) p[i] = v;
}

__global__ void relu_inplace_kernel(float* __restrict__ p, long long n) {
    long long i = (long long)blockIdx.x * blockDim.x + threadIdx.x;
    if (i < n) p[i] = fmaxf(p[i], 0.0f);
}

__global__ void rsqrt_inplace_kernel(float* __restrict__ p, int n) {
    int i = blockIdx.x * blockDim.x + threadIdx.x;
    if (i < n) p[i] = rsqrtf(p[i]);
}

// out[n*K + k] = (f16) W[k*N + n]   (transposed f16 weights, [N,K])
__global__ void cvt_t_f16_kernel(const float* __restrict__ W,
                                 _Float16* __restrict__ out, int K, int N) {
    int idx = blockIdx.x * blockDim.x + threadIdx.x;
    if (idx >= K * N) return;
    int n = idx / K, k = idx % K;
    out[idx] = (_Float16)W[(size_t)k * N + n];
}

// plain f32 -> f16 (for W_ih / W_hh which are already [N,K] for x @ W.T)
__global__ void cvt_f16_kernel(const float* __restrict__ W,
                               _Float16* __restrict__ out, int n) {
    int i = blockIdx.x * blockDim.x + threadIdx.x;
    if (i < n) out[i] = (_Float16)W[i];
}

// deg[dst[e]] += 1
__global__ void deg_edges_kernel(const int* __restrict__ dst,
                                 float* __restrict__ deg, int E) {
    int e = blockIdx.x * blockDim.x + threadIdx.x;
    if (e < E) atomicAdd(&deg[dst[e]], 1.0f);
}

// ---------------------------------------------------------------------------
// WMMA GEMM: C[M,N] = A[M,K] (f32) x Bt^T (Bt is [N,K] f16) + bias, opt ReLU
// Each wave computes a 64x16 block of C (4 WMMA tiles sharing one B fragment).
// Requires M % 64 == 0, N % 16 == 0, K % 32 == 0.
// ---------------------------------------------------------------------------
__global__ void gemm_wmma_kernel(const float* __restrict__ A,
                                 const _Float16* __restrict__ Bt,
                                 const float* __restrict__ bias,
                                 float* __restrict__ C,
                                 int M, int N, int K, int relu) {
    const int wave = threadIdx.x >> 5;
    const int lane = threadIdx.x & 31;
    const int tilesN = N >> 4;
    const int tilesM = M >> 6;
    const int tile = blockIdx.x * (blockDim.x >> 5) + wave;
    if (tile >= tilesM * tilesN) return;
    const int tm = tile / tilesN;
    const int tn = tile % tilesN;

    const int l15 = lane & 15;
    const int hi  = lane >> 4;     // half-wave selector per ISA lane layout

    v8f acc[4];
    for (int s = 0; s < 4; ++s)
        for (int r = 0; r < 8; ++r) acc[s][r] = 0.0f;

    const int col = tn * 16 + l15;
    const _Float16* __restrict__ bcol = Bt + (size_t)col * K;

    for (int k0 = 0; k0 < K; k0 += 32) {
        // B fragment: lane (col, hi) holds K = k0 + hi*16 .. +15, contiguous.
        v16h bfrag = *reinterpret_cast<const v16h*>(bcol + k0 + hi * 16);
        const int kA = k0 + hi * 16;
        for (int s = 0; s < 4; ++s) {
            const int row = (tm << 6) + (s << 4) + l15;
            const float* ap = A + (size_t)row * K + kA;
            const float4 a0 = reinterpret_cast<const float4*>(ap)[0];
            const float4 a1 = reinterpret_cast<const float4*>(ap)[1];
            const float4 a2 = reinterpret_cast<const float4*>(ap + 8)[0];
            const float4 a3 = reinterpret_cast<const float4*>(ap + 8)[1];
            v16h af;
            af[0]=(_Float16)a0.x;  af[1]=(_Float16)a0.y;
            af[2]=(_Float16)a0.z;  af[3]=(_Float16)a0.w;
            af[4]=(_Float16)a1.x;  af[5]=(_Float16)a1.y;
            af[6]=(_Float16)a1.z;  af[7]=(_Float16)a1.w;
            af[8]=(_Float16)a2.x;  af[9]=(_Float16)a2.y;
            af[10]=(_Float16)a2.z; af[11]=(_Float16)a2.w;
            af[12]=(_Float16)a3.x; af[13]=(_Float16)a3.y;
            af[14]=(_Float16)a3.z; af[15]=(_Float16)a3.w;
            acc[s] = __builtin_amdgcn_wmma_f32_16x16x32_f16(
                false, af, false, bfrag, (short)0, acc[s], false, false);
        }
    }

    const float bv = bias ? bias[col] : 0.0f;
    for (int s = 0; s < 4; ++s) {
        const int rowBase = (tm << 6) + (s << 4) + hi * 8;
        for (int r = 0; r < 8; ++r) {
            float v = acc[s][r] + bv;
            if (relu) v = fmaxf(v, 0.0f);
            C[(size_t)(rowBase + r) * N + col] = v;
        }
    }
}

// ---------------------------------------------------------------------------
// GCN aggregation: out = self-loop term + bias, then edge scatter, then ReLU
// ---------------------------------------------------------------------------
__global__ void gcn_self_init_kernel(const float* __restrict__ xw,
                                     const float* __restrict__ dinv,
                                     const float* __restrict__ bias,
                                     float* __restrict__ out,
                                     long long total, int C) {
    long long idx = (long long)blockIdx.x * blockDim.x + threadIdx.x;
    if (idx >= total) return;
    int i = (int)(idx / C), c = (int)(idx % C);
    float di = dinv[i];
    out[idx] = xw[idx] * di * di + bias[c];
}

__global__ void gcn_scatter_kernel(const int* __restrict__ src,
                                   const int* __restrict__ dst,
                                   const float* __restrict__ xw,
                                   const float* __restrict__ dinv,
                                   float* __restrict__ out,
                                   long long total, int C) {
    long long idx = (long long)blockIdx.x * blockDim.x + threadIdx.x;
    if (idx >= total) return;
    int e = (int)(idx / C), c = (int)(idx % C);
    int s = src[e], d = dst[e];
    float v = xw[(size_t)s * C + c] * dinv[s] * dinv[d];
    atomicAdd(&out[(size_t)d * C + c], v);
}

// ---------------------------------------------------------------------------
// Attention pooling over contiguous segments. blockIdx = segment,
// threadIdx = channel (C == blockDim.x == 128).
// ---------------------------------------------------------------------------
__global__ void att_pool_kernel(const float* __restrict__ x,
                                const float* __restrict__ gate,
                                float* __restrict__ out,
                                int seglen, int C) {
    int seg = blockIdx.x;
    int c = threadIdx.x;
    size_t base = (size_t)seg * seglen;
    float m = -3.4e38f;
    for (int i = 0; i < seglen; ++i)
        m = fmaxf(m, gate[(base + i) * C + c]);
    float s = 0.0f, acc = 0.0f;
    for (int i = 0; i < seglen; ++i) {
        float e = expf(gate[(base + i) * C + c] - m);
        s += e;
        acc += e * x[(base + i) * C + c];
    }
    out[(size_t)seg * C + c] = acc / (s + 1e-16f);
}

// ---------------------------------------------------------------------------
// GRU element-wise step (gi precomputed for all t; gh precomputed per step)
// ---------------------------------------------------------------------------
__global__ void gru_step_kernel(const float* __restrict__ GI,
                                const float* __restrict__ GH,
                                float* __restrict__ h, int t) {
    int idx = blockIdx.x * blockDim.x + threadIdx.x;   // b*128 + c
    if (idx >= BB * 128) return;
    int b = idx >> 7, c = idx & 127;
    const float* gi = GI + (size_t)(b * TT + t) * 384;
    const float* gh = GH + (size_t)b * 384;
    float r  = 1.0f / (1.0f + expf(-(gi[c]       + gh[c])));
    float z  = 1.0f / (1.0f + expf(-(gi[128 + c] + gh[128 + c])));
    float n  = tanhf(gi[256 + c] + r * gh[256 + c]);
    h[idx] = (1.0f - z) * n + z * h[idx];
}

// ---------------------------------------------------------------------------
// Head: concat [user_feats | user_pool | tweet_h], final dot + sigmoid
// ---------------------------------------------------------------------------
__global__ void concat_head_kernel(const float* __restrict__ uf,
                                   const float* __restrict__ up,
                                   const float* __restrict__ th,
                                   float* __restrict__ full) {
    int idx = blockIdx.x * blockDim.x + threadIdx.x;
    if (idx >= BB * 288) return;
    int b = idx / 288, c = idx % 288;
    float v;
    if (c < 32)       v = uf[b * 32 + c];
    else if (c < 160) v = up[b * 128 + (c - 32)];
    else              v = th[b * 128 + (c - 160)];
    full[idx] = v;
}

__global__ void head_final_kernel(const float* __restrict__ X,
                                  const float* __restrict__ Wp,
                                  const float* __restrict__ bp,
                                  float* __restrict__ out) {
    int b = threadIdx.x;
    if (b >= BB) return;
    float s = bp[0];
    for (int k = 0; k < 64; ++k) s += X[b * 64 + k] * Wp[k];
    out[b] = 1.0f / (1.0f + expf(-s));
}

// ---------------------------------------------------------------------------
// Launch
// ---------------------------------------------------------------------------
static inline unsigned nblk(long long n, int t) {
    return (unsigned)((n + t - 1) / t);
}

extern "C" void kernel_launch(void* const* d_in, const int* in_sizes, int n_in,
                              void* d_out, int out_size, void* d_ws, size_t ws_size,
                              hipStream_t stream) {
    (void)in_sizes; (void)n_in; (void)out_size; (void)ws_size;

    const float* user_x     = (const float*)d_in[0];
    const float* user_feats = (const float*)d_in[1];
    const float* tweet_x    = (const float*)d_in[2];
    const float* Wu0 = (const float*)d_in[3];  const float* bu0 = (const float*)d_in[4];
    const float* Wu1 = (const float*)d_in[5];  const float* bu1 = (const float*)d_in[6];
    const float* Wgu1= (const float*)d_in[7];  const float* bgu1= (const float*)d_in[8];
    const float* Wgu2= (const float*)d_in[9];  const float* bgu2= (const float*)d_in[10];
    const float* Wt0 = (const float*)d_in[11]; const float* bt0 = (const float*)d_in[12];
    const float* Wt1 = (const float*)d_in[13]; const float* bt1 = (const float*)d_in[14];
    const float* Wgt1= (const float*)d_in[15]; const float* bgt1= (const float*)d_in[16];
    const float* Wgt2= (const float*)d_in[17]; const float* bgt2= (const float*)d_in[18];
    const float* W_ih= (const float*)d_in[19]; const float* b_ih= (const float*)d_in[20];
    const float* W_hh= (const float*)d_in[21]; const float* b_hh= (const float*)d_in[22];
    const float* Wd0 = (const float*)d_in[23]; const float* bd0 = (const float*)d_in[24];
    const float* Wd1 = (const float*)d_in[25]; const float* bd1 = (const float*)d_in[26];
    const float* Wp  = (const float*)d_in[27]; const float* bp  = (const float*)d_in[28];
    const int* uei = (const int*)d_in[29];           // [2, EU]
    const int* tei = (const int*)d_in[30];           // [2, NTE]
    const int* srcU = uei,          *dstU = uei + EU;
    const int* srcT = tei,          *dstT = tei + NTE;

    // ---- workspace bump allocator -------------------------------------
    uint8_t* base = (uint8_t*)d_ws;
    size_t off = 0;
    auto alloc = [&](size_t bytes) -> void* {
        void* p = base + off;
        off = (off + bytes + 255) & ~(size_t)255;
        return p;
    };

    _Float16* wu0t  = (_Float16*)alloc(128 * 128 * 2);
    _Float16* wu1t  = (_Float16*)alloc(128 * 128 * 2);
    _Float16* wgu1t = (_Float16*)alloc(128 * 128 * 2);
    _Float16* wgu2t = (_Float16*)alloc(128 * 128 * 2);
    _Float16* wt0t  = (_Float16*)alloc(768 * 256 * 2);
    _Float16* wt1t  = (_Float16*)alloc(256 * 128 * 2);
    _Float16* wgt1t = (_Float16*)alloc(128 * 128 * 2);
    _Float16* wgt2t = (_Float16*)alloc(128 * 128 * 2);
    _Float16* wiht  = (_Float16*)alloc(384 * 128 * 2);
    _Float16* whht  = (_Float16*)alloc(384 * 128 * 2);
    _Float16* wd0t  = (_Float16*)alloc(288 * 256 * 2);
    _Float16* wd1t  = (_Float16*)alloc(256 * 64 * 2);

    float* dinvU = (float*)alloc((size_t)NU * 4);
    float* dinvT = (float*)alloc((size_t)NT * 4);
    float* T0 = (float*)alloc((size_t)NT * 256 * 4);   // 31.5 MB, also fits NUx128
    float* T1 = (float*)alloc((size_t)NT * 256 * 4);   // 31.5 MB
    float* G  = (float*)alloc((size_t)NU * 128 * 4);   // 20.5 MB (gate tmp)
    float* G2 = (float*)alloc((size_t)NU * 128 * 4);   // 20.5 MB (gate)
    float* user_pool  = (float*)alloc((size_t)BB * 128 * 4);
    float* tweet_pool = (float*)alloc((size_t)NSEG * 128 * 4);
    float* GI   = (float*)alloc((size_t)NSEG * 384 * 4);
    float* GH   = (float*)alloc((size_t)BB * 384 * 4);
    float* hbuf = (float*)alloc((size_t)BB * 128 * 4);
    float* full = (float*)alloc((size_t)BB * 288 * 4);
    float* hd1  = (float*)alloc((size_t)BB * 256 * 4);
    float* hd2  = (float*)alloc((size_t)BB * 64 * 4);

    // ---- weight conversion (f16, transposed to [N,K]) -----------------
    auto cvtT = [&](const float* W, _Float16* o, int K, int N) {
        cvt_t_f16_kernel<<<nblk((long long)K * N, 256), 256, 0, stream>>>(W, o, K, N);
    };
    cvtT(Wu0,  wu0t,  128, 128);  cvtT(Wu1,  wu1t,  128, 128);
    cvtT(Wgu1, wgu1t, 128, 128);  cvtT(Wgu2, wgu2t, 128, 128);
    cvtT(Wt0,  wt0t,  768, 256);  cvtT(Wt1,  wt1t,  256, 128);
    cvtT(Wgt1, wgt1t, 128, 128);  cvtT(Wgt2, wgt2t, 128, 128);
    cvtT(Wd0,  wd0t,  288, 256);  cvtT(Wd1,  wd1t,  256, 64);
    cvt_f16_kernel<<<nblk(384 * 128, 256), 256, 0, stream>>>(W_ih, wiht, 384 * 128);
    cvt_f16_kernel<<<nblk(384 * 128, 256), 256, 0, stream>>>(W_hh, whht, 384 * 128);

    // ---- degree normalization (deg = indeg + self loop) ---------------
    fill_f32_kernel<<<nblk(NU, 256), 256, 0, stream>>>(dinvU, 1.0f, NU);
    deg_edges_kernel<<<nblk(EU, 256), 256, 0, stream>>>(dstU, dinvU, EU);
    rsqrt_inplace_kernel<<<nblk(NU, 256), 256, 0, stream>>>(dinvU, NU);
    fill_f32_kernel<<<nblk(NT, 256), 256, 0, stream>>>(dinvT, 1.0f, NT);
    deg_edges_kernel<<<nblk(NTE, 256), 256, 0, stream>>>(dstT, dinvT, NTE);
    rsqrt_inplace_kernel<<<nblk(NT, 256), 256, 0, stream>>>(dinvT, NT);

    auto gemm = [&](const float* A, const _Float16* Bt, const float* bias,
                    float* C, int M, int N, int K, int relu) {
        long long tiles = (long long)(M >> 6) * (N >> 4);
        gemm_wmma_kernel<<<nblk(tiles, 4), 128, 0, stream>>>(A, Bt, bias, C, M, N, K, relu);
    };
    auto gcn_agg = [&](const float* xw, const float* dinv, const float* bias,
                       float* out, const int* src, const int* dst,
                       int Nn, int E, int C) {
        long long tn = (long long)Nn * C;
        long long te = (long long)E * C;
        gcn_self_init_kernel<<<nblk(tn, 256), 256, 0, stream>>>(xw, dinv, bias, out, tn, C);
        gcn_scatter_kernel<<<nblk(te, 256), 256, 0, stream>>>(src, dst, xw, dinv, out, te, C);
        relu_inplace_kernel<<<nblk(tn, 256), 256, 0, stream>>>(out, tn);
    };

    // ---- user branch: 2x GCNConv + attention pooling ------------------
    gemm(user_x, wu0t, nullptr, T0, NU, 128, 128, 0);
    gcn_agg(T0, dinvU, bu0, T1, srcU, dstU, NU, EU, 128);
    gemm(T1, wu1t, nullptr, T0, NU, 128, 128, 0);
    gcn_agg(T0, dinvU, bu1, T1, srcU, dstU, NU, EU, 128);      // T1 = h2
    gemm(T1, wgu1t, bgu1, G,  NU, 128, 128, 0);
    gemm(G,  wgu2t, bgu2, G2, NU, 128, 128, 0);
    att_pool_kernel<<<BB, 128, 0, stream>>>(T1, G2, user_pool, NU / BB, 128);

    // ---- tweet branch: 2x GCNConv + attention pooling -----------------
    gemm(tweet_x, wt0t, nullptr, T0, NT, 256, 768, 0);
    gcn_agg(T0, dinvT, bt0, T1, srcT, dstT, NT, NTE, 256);
    gemm(T1, wt1t, nullptr, T0, NT, 128, 256, 0);
    gcn_agg(T0, dinvT, bt1, T1, srcT, dstT, NT, NTE, 128);     // T1 = g2
    gemm(T1, wgt1t, bgt1, G,  NT, 128, 128, 0);
    gemm(G,  wgt2t, bgt2, G2, NT, 128, 128, 0);
    att_pool_kernel<<<NSEG, 128, 0, stream>>>(T1, G2, tweet_pool, NW, 128);

    // ---- GRU over tweets ----------------------------------------------
    gemm(tweet_pool, wiht, b_ih, GI, NSEG, 384, 128, 0);       // all gi up front
    fill_f32_kernel<<<nblk(BB * 128, 256), 256, 0, stream>>>(hbuf, 0.0f, BB * 128);
    for (int t = 0; t < TT; ++t) {
        gemm(hbuf, whht, b_hh, GH, BB, 384, 128, 0);
        gru_step_kernel<<<nblk(BB * 128, 256), 256, 0, stream>>>(GI, GH, hbuf, t);
    }

    // ---- head ----------------------------------------------------------
    concat_head_kernel<<<nblk(BB * 288, 256), 256, 0, stream>>>(user_feats, user_pool,
                                                                hbuf, full);
    gemm(full, wd0t, bd0, hd1, BB, 256, 288, 1);
    gemm(hd1,  wd1t, bd1, hd2, BB, 64, 256, 1);
    head_final_kernel<<<1, 64, 0, stream>>>(hd2, Wp, bp, (float*)d_out);
}
